// MincutPool_43104291783363
// MI455X (gfx1250) — matrix operand order repaired and checked by reading, not compiled, not run on hardware
//
#include <hip/hip_runtime.h>
#include <math.h>

typedef __attribute__((ext_vector_type(16))) _Float16 v16h;
typedef __attribute__((ext_vector_type(8)))  _Float16 v8h;
typedef __attribute__((ext_vector_type(4)))  _Float16 v4h;
typedef __attribute__((ext_vector_type(8)))  float    v8f;

#define B_GRAPHS 8
#define NPG      4096      // nodes per graph
#define KCLUS    64        // clusters
#define FDIM     128       // feature dim (all layers)

__device__ __forceinline__ v8f wmma_f16(v16h a, v16h b, v8f c) {
    return __builtin_amdgcn_wmma_f32_16x16x32_f16(false, a, false, b, (short)0, c, false, false);
}

__device__ __forceinline__ v16h cat8(v8h lo, v8h hi) {
    return __builtin_shufflevector(lo, hi, 0, 1, 2, 3, 4, 5, 6, 7,
                                           8, 9, 10, 11, 12, 13, 14, 15);
}

// ---------------------------------------------------------------------------
// Pack f32 [Krows, N] row-major -> f16 WMMA-B layout:
//   Bp[((kc*colTiles + c)*32 + lane)*16 + e] = B[(kc*32 + (lane>>4)*16 + e)*N + c*16 + (lane&15)]
// One thread per (kc, c, lane): 16 strided reads, one 32B v16h store.
// ---------------------------------------------------------------------------
__global__ void mcp_pack_b(const float* __restrict__ Bsrc, _Float16* __restrict__ Bp,
                           int Krows, int N) {
    const int tid = blockIdx.x * blockDim.x + threadIdx.x;
    const int colTiles = N >> 4;
    const int total = (Krows >> 5) * colTiles * 32;
    if (tid >= total) return;
    const int lane = tid & 31;
    const int c  = (tid >> 5) % colTiles;
    const int kc = (tid >> 5) / colTiles;
    const int g = lane >> 4, r = lane & 15;
    const float* srcp = Bsrc + (size_t)(kc * 32 + g * 16) * N + c * 16 + r;
    v16h v;
#pragma unroll
    for (int e = 0; e < 16; ++e) v[e] = (_Float16)srcp[(size_t)e * N];
    ((v16h*)Bp)[tid] = v;
}

// f32 -> f16 elementwise (4 elems/thread)
__global__ void mcp_cvt_f16(const float* __restrict__ in, _Float16* __restrict__ outh,
                            int nElem) {
    const int i = blockIdx.x * blockDim.x + threadIdx.x;
    if (i * 4 >= nElem) return;
    float4 v = ((const float4*)in)[i];
    v4h o = { (_Float16)v.x, (_Float16)v.y, (_Float16)v.z, (_Float16)v.w };
    ((v4h*)outh)[i] = o;
}

// ---------------------------------------------------------------------------
// C[M,N] = act(A[M,128] @ W[128,N] + bias).  Ah: f16 row-major.  Bp: packed f16.
// act: 0 none, 1 relu, 2 tanh.  Ch (optional): f16 row-major copy of C.
// One wave per 16x16 tile; A fetch = two v8h loads, B fetch = one v16h load.
// ---------------------------------------------------------------------------
__global__ void mcp_gemm_xw(const _Float16* __restrict__ Ah, const _Float16* __restrict__ Bp,
                            const float* __restrict__ bias, float* __restrict__ C,
                            _Float16* __restrict__ Ch, int M, int N, int act) {
    const int lane = threadIdx.x & 31;
    const int wave = threadIdx.x >> 5;
    const int colTiles = N >> 4;
    const int tile = blockIdx.x * (blockDim.x >> 5) + wave;
    if (tile >= (M >> 4) * colTiles) return;          // wave-uniform exit
    const int rowTile = tile / colTiles;
    const int colTile = tile % colTiles;
    const int r = lane & 15;
    const int g = lane >> 4;
    const int row0 = rowTile * 16;
    const int col0 = colTile * 16;

    const _Float16* arow = Ah + (size_t)(row0 + r) * FDIM + g * 16;
    const v16h* bp = (const v16h*)Bp;
    v8f acc = {};
#pragma unroll
    for (int kc = 0; kc < FDIM / 32; ++kc) {
        v8h lo = *(const v8h*)(arow + kc * 32);        // k = kc*32 + g*16 + [0,8)
        v8h hi = *(const v8h*)(arow + kc * 32 + 8);    // k = kc*32 + g*16 + 8 + [0,8)
        v16h a = cat8(lo, hi);
        v16h b = bp[((size_t)kc * colTiles + colTile) * 32 + lane];
        acc = wmma_f16(a, b, acc);
    }
    const float bv = bias ? bias[col0 + r] : 0.0f;
    const int rowOut0 = row0 + g * 8;                  // D: elem j -> row +j, col r
#pragma unroll
    for (int j = 0; j < 8; ++j) {
        float v = acc[j] + bv;
        if (act == 1)      v = fmaxf(v, 0.0f);
        else if (act == 2) v = tanhf(v);
        const size_t idx = (size_t)(rowOut0 + j) * N + col0 + r;
        C[idx] = v;
        if (Ch) Ch[idx] = (_Float16)v;
    }
}

// ---------------------------------------------------------------------------
// Per-batch C[b][64,N] = s_b^T @ B_b.  At: f16 s^T [B][64][4096] row-major.
// Bp: packed f16 B layout over all B*4096 rows. K = 4096 per batch.
// ---------------------------------------------------------------------------
__global__ void mcp_gemm_atb(const _Float16* __restrict__ At, const _Float16* __restrict__ Bp,
                             float* __restrict__ C, _Float16* __restrict__ Ch, int N) {
    const int lane = threadIdx.x & 31;
    const int wave = threadIdx.x >> 5;
    const int colTiles = N >> 4;
    const int tilesPerBatch = (KCLUS >> 4) * colTiles;
    const int tile = blockIdx.x * (blockDim.x >> 5) + wave;
    if (tile >= B_GRAPHS * tilesPerBatch) return;      // wave-uniform exit
    const int b = tile / tilesPerBatch;
    const int t = tile % tilesPerBatch;
    const int rowTile = t / colTiles;
    const int colTile = t % colTiles;
    const int r = lane & 15;
    const int g = lane >> 4;
    const int m   = rowTile * 16 + r;                  // cluster index (C row)
    const int col = colTile * 16 + r;

    const _Float16* arow = At + ((size_t)b * KCLUS + m) * NPG + g * 16;
    const v16h* bp = (const v16h*)Bp;
    v8f acc = {};
    for (int kc = 0; kc < NPG / 32; ++kc) {
        v8h lo = *(const v8h*)(arow + kc * 32);
        v8h hi = *(const v8h*)(arow + kc * 32 + 8);
        v16h a = cat8(lo, hi);
        v16h bb = bp[(((size_t)b * (NPG / 32) + kc) * colTiles + colTile) * 32 + lane];
        acc = wmma_f16(a, bb, acc);
    }
    const int rowOut0 = rowTile * 16 + g * 8;
#pragma unroll
    for (int j = 0; j < 8; ++j) {
        const size_t idx = ((size_t)b * KCLUS + rowOut0 + j) * N + col;
        C[idx] = acc[j];
        if (Ch) Ch[idx] = (_Float16)acc[j];
    }
}

// ---------------------------------------------------------------------------
// Degree / scatter / misc kernels (bandwidth-bound; edge waves use scalar f32 atomics)
// ---------------------------------------------------------------------------
__global__ void mcp_deg_init(float* deg, float* outdeg, int n) {
    int i = blockIdx.x * blockDim.x + threadIdx.x;
    if (i < n) { deg[i] = 1.0f; outdeg[i] = 0.0f; }    // self-loop counts as 1
}

__global__ void mcp_deg_count(const int* __restrict__ src, const int* __restrict__ dst,
                              int ne, float* deg, float* outdeg) {
    int e = blockIdx.x * blockDim.x + threadIdx.x;
    if (e < ne) {
        atomicAdd(&deg[dst[e]], 1.0f);     // GCN degree (in-degree + self loop)
        atomicAdd(&outdeg[src[e]], 1.0f);  // d_flat = adj.sum(-1) (out-degree)
    }
}

__global__ void mcp_deg_rsqrt(float* deg, int n) {
    int i = blockIdx.x * blockDim.x + threadIdx.x;
    if (i < n) deg[i] = rsqrtf(deg[i]);    // becomes dis[]
}

// h[n] = hw[n] * dis[n]^2   (self-loop message, initializes h)
__global__ void mcp_selfloop(const float* __restrict__ hw, const float* __restrict__ dis,
                             float* __restrict__ h, int nNodes) {
    int i = blockIdx.x * blockDim.x + threadIdx.x;
    if (i < nNodes * FDIM) {
        int n = i >> 7;
        float d = dis[n];
        h[i] = hw[i] * d * d;
    }
}

// wave per edge: h[dst] += dis[src]*dis[dst] * hw[src]
__global__ void mcp_edge_scatter(const int* __restrict__ src, const int* __restrict__ dst,
                                 const float* __restrict__ dis, const float* __restrict__ hw,
                                 float* __restrict__ h, int ne) {
    int w = (blockIdx.x * blockDim.x + threadIdx.x) >> 5;
    int lane = threadIdx.x & 31;
    if (w >= ne) return;
    int s = src[w], d = dst[w];
    float c = dis[s] * dis[d];
    float4 v = ((const float4*)(hw + (size_t)s * FDIM))[lane];
    float* hd = h + (size_t)d * FDIM + lane * 4;
    atomicAdd(hd + 0, c * v.x);
    atomicAdd(hd + 1, c * v.y);
    atomicAdd(hd + 2, c * v.z);
    atomicAdd(hd + 3, c * v.w);
}

// bias + optional relu; also emits f16 copy for the next GEMM's A operand
__global__ void mcp_bias_act(float* __restrict__ h, const float* __restrict__ bias,
                             _Float16* __restrict__ hh, int nNodes, int relu) {
    int i = blockIdx.x * blockDim.x + threadIdx.x;
    if (i < nNodes * FDIM) {
        float v = h[i] + bias[i & (FDIM - 1)];
        if (relu) v = fmaxf(v, 0.0f);
        h[i] = v;
        if (hh) hh[i] = (_Float16)v;
    }
}

// wave per node, 64 values (lane*2), softmax applied twice (faithful to reference)
__global__ void mcp_softmax2(float* __restrict__ s, int nNodes) {
    int w = (blockIdx.x * blockDim.x + threadIdx.x) >> 5;
    int lane = threadIdx.x & 31;
    if (w >= nNodes) return;
    float2 v = ((float2*)(s + (size_t)w * KCLUS))[lane];
    for (int pass = 0; pass < 2; ++pass) {
        float mx = fmaxf(v.x, v.y);
        for (int o = 16; o > 0; o >>= 1) mx = fmaxf(mx, __shfl_xor(mx, o, 32));
        float ex = __expf(v.x - mx), ey = __expf(v.y - mx);
        float sm = ex + ey;
        for (int o = 16; o > 0; o >>= 1) sm += __shfl_xor(sm, o, 32);
        v.x = ex / sm; v.y = ey / sm;
    }
    ((float2*)(s + (size_t)w * KCLUS))[lane] = v;
}

// s [B*4096, 64] f32 -> s^T [B][64][4096] f16 (coalesced writes)
__global__ void mcp_transpose_s(const float* __restrict__ s, _Float16* __restrict__ sT,
                                int nNodes) {
    int i = blockIdx.x * blockDim.x + threadIdx.x;
    if (i >= nNodes * KCLUS) return;
    int n = i & (NPG - 1);
    int k = (i >> 12) & (KCLUS - 1);
    int b = i >> 18;
    sT[i] = (_Float16)s[((size_t)b * NPG + n) * KCLUS + k];
}

// wave per edge: As[src] += s_d[dst]   (sparse A @ s, no dense [B,4096,4096])
__global__ void mcp_as_scatter(const int* __restrict__ src, const int* __restrict__ dst,
                               const float* __restrict__ s, float* __restrict__ As, int ne) {
    int w = (blockIdx.x * blockDim.x + threadIdx.x) >> 5;
    int lane = threadIdx.x & 31;
    if (w >= ne) return;
    int sn = src[w], dn = dst[w];
    float2 v = ((const float2*)(s + (size_t)dn * KCLUS))[lane];
    float* p = As + (size_t)sn * KCLUS + lane * 2;
    atomicAdd(p + 0, v.x);
    atomicAdd(p + 1, v.y);
}

// mincut_den[b] = sum_n outdeg[n] * sum_k s[n,k]^2
__global__ void mcp_den(const float* __restrict__ s, const float* __restrict__ outdeg,
                        float* __restrict__ den, int nNodes) {
    int w = (blockIdx.x * blockDim.x + threadIdx.x) >> 5;
    int lane = threadIdx.x & 31;
    if (w >= nNodes) return;
    float2 v = ((const float2*)(s + (size_t)w * KCLUS))[lane];
    float val = v.x * v.x + v.y * v.y;
    for (int o = 16; o > 0; o >>= 1) val += __shfl_xor(val, o, 32);
    if (lane == 0) atomicAdd(&den[w / NPG], outdeg[w] * val);
}

__global__ void mcp_init_misc(float* den, float* dout, int offBatch, int offLoss) {
    int t = blockIdx.x * blockDim.x + threadIdx.x;
    if (t < B_GRAPHS) den[t] = 0.0f;
    if (t < B_GRAPHS * KCLUS) dout[offBatch + t] = (float)(t >> 6);
    if (t == 0) { dout[offLoss] = 0.0f; dout[offLoss + 1] = 0.0f; }
}

// one block (256 thr) per batch: trace, mincut loss, ortho loss, adj normalization
__global__ void mcp_finalize(const float* __restrict__ ssM, const float* __restrict__ adjM,
                             const float* __restrict__ den, float* __restrict__ outAdj,
                             float* __restrict__ lossPtr /* [mincut, ortho] */) {
    __shared__ float sm[KCLUS * KCLUS];
    __shared__ float red[256];
    __shared__ float dd[KCLUS];
    const int b = blockIdx.x, t = threadIdx.x;
    const float invB = 1.0f / (float)B_GRAPHS;

    const float* ssb = ssM + (size_t)b * KCLUS * KCLUS;
    float acc = 0.0f;
    for (int i = t; i < KCLUS * KCLUS; i += 256) { float v = ssb[i]; sm[i] = v; acc += v * v; }
    red[t] = acc; __syncthreads();
    for (int s2 = 128; s2 > 0; s2 >>= 1) { if (t < s2) red[t] += red[t + s2]; __syncthreads(); }
    const float ssn = sqrtf(red[0]);
    __syncthreads();

    const float inv = 1.0f / ssn, diag = 1.0f / sqrtf((float)KCLUS);
    acc = 0.0f;
    for (int i = t; i < KCLUS * KCLUS; i += 256) {
        int kk = i >> 6, jj = i & 63;
        float v = sm[i] * inv - (kk == jj ? diag : 0.0f);
        acc += v * v;
    }
    red[t] = acc; __syncthreads();
    for (int s2 = 128; s2 > 0; s2 >>= 1) { if (t < s2) red[t] += red[t + s2]; __syncthreads(); }
    if (t == 0) atomicAdd(&lossPtr[1], sqrtf(red[0]) * invB);
    __syncthreads();

    const float* ab = adjM + (size_t)b * KCLUS * KCLUS;
    acc = 0.0f;
    for (int i = t; i < KCLUS * KCLUS; i += 256) {
        float v = ab[i]; int kk = i >> 6, jj = i & 63;
        if (kk == jj) { acc += v; v = 0.0f; }          // trace before zeroing diag
        sm[i] = v;
    }
    red[t] = acc; __syncthreads();
    for (int s2 = 128; s2 > 0; s2 >>= 1) { if (t < s2) red[t] += red[t + s2]; __syncthreads(); }
    if (t == 0) atomicAdd(&lossPtr[0], -(red[0] / den[b]) * invB);
    __syncthreads();
    if (t < KCLUS) {
        float rs = 0.0f;
        for (int j = 0; j < KCLUS; ++j) rs += sm[t * KCLUS + j];
        dd[t] = sqrtf(rs) + 1e-15f;
    }
    __syncthreads();
    for (int i = t; i < KCLUS * KCLUS; i += 256) {
        int kk = i >> 6, jj = i & 63;
        outAdj[(size_t)b * KCLUS * KCLUS + i] = sm[i] / (dd[kk] * dd[jj]);
    }
}

// ---------------------------------------------------------------------------
extern "C" void kernel_launch(void* const* d_in, const int* in_sizes, int n_in,
                              void* d_out, int out_size, void* d_ws, size_t ws_size,
                              hipStream_t stream) {
    const float* x   = (const float*)d_in[0];
    const int*   ei  = (const int*)  d_in[1];
    const float* W1  = (const float*)d_in[3];
    const float* b1  = (const float*)d_in[4];
    const float* W2  = (const float*)d_in[5];
    const float* b2  = (const float*)d_in[6];
    const float* Wa1 = (const float*)d_in[7];
    const float* ba1 = (const float*)d_in[8];
    const float* Wa2 = (const float*)d_in[9];
    const float* ba2 = (const float*)d_in[10];
    const float* Wo  = (const float*)d_in[11];
    const float* bo  = (const float*)d_in[12];
    float* out = (float*)d_out;

    const int NODES = in_sizes[0] / FDIM;     // 32768
    const int NE    = in_sizes[1] / 2;        // 524288
    const int* src = ei;
    const int* dst = ei + NE;

    // ---- workspace carve: f32 region then f16 region (32B aligned) ----
    float* ws = (float*)d_ws;
    size_t off = 0;
    float* hw     = ws + off; off += (size_t)NODES * FDIM;
    float* h1     = ws + off; off += (size_t)NODES * FDIM;
    float* h2     = ws + off; off += (size_t)NODES * FDIM;
    float* s      = ws + off; off += (size_t)NODES * KCLUS;
    float* As     = ws + off; off += (size_t)NODES * KCLUS;
    float* dis    = ws + off; off += NODES;
    float* outdeg = ws + off; off += NODES;
    float* poolX  = ws + off; off += (size_t)B_GRAPHS * KCLUS * FDIM;
    float* adjRaw = ws + off; off += (size_t)B_GRAPHS * KCLUS * KCLUS;
    float* ssRaw  = ws + off; off += (size_t)B_GRAPHS * KCLUS * KCLUS;
    float* den    = ws + off; off += 16;                       // 8 used, pad to 64B

    _Float16* hp = (_Float16*)(ws + off);
    size_t ho = 0;
    _Float16* F16A   = hp + ho; ho += (size_t)NODES * FDIM;    // xh -> th -> h2p
    _Float16* F16B   = hp + ho; ho += (size_t)NODES * FDIM;    // h1h -> h2h
    _Float16* sp     = hp + ho; ho += (size_t)NODES * KCLUS;   // s packed (B layout)
    _Float16* Asp    = hp + ho; ho += (size_t)NODES * KCLUS;   // As packed (B layout)
    _Float16* sTh    = hp + ho; ho += (size_t)NODES * KCLUS;   // s^T f16
    _Float16* poolXh = hp + ho; ho += (size_t)B_GRAPHS * KCLUS * FDIM;
    _Float16* W1p    = hp + ho; ho += 4 * 8 * 32 * 16;
    _Float16* W2p    = hp + ho; ho += 4 * 8 * 32 * 16;
    _Float16* Wa1p   = hp + ho; ho += 4 * 8 * 32 * 16;
    _Float16* Wa2p   = hp + ho; ho += 4 * 4 * 32 * 16;
    _Float16* Wop    = hp + ho; ho += 4 * 8 * 32 * 16;

    // d_out layout: pooled_x | out_adj | pooled_batch | mincut | ortho
    const int offAdj   = B_GRAPHS * KCLUS * FDIM;
    const int offBatch = offAdj + B_GRAPHS * KCLUS * KCLUS;
    const int offLoss  = offBatch + B_GRAPHS * KCLUS;

    const int T = 256;
    const int elemBlocks = (NODES * FDIM + T - 1) / T;
    const int waveEdgeBlocks = (NE * 32 + T - 1) / T;
    const int waveNodeBlocks = (NODES * 32 + T - 1) / T;
    auto gemmBlocks = [](int M, int N) { return ((M >> 4) * (N >> 4) + 7) / 8; };
    auto packBlocks = [T](int K, int N) { return ((K >> 5) * (N >> 4) * 32 + T - 1) / T; };

    // ---- init + degrees ----
    mcp_init_misc<<<2, T, 0, stream>>>(den, out, offBatch, offLoss);
    mcp_deg_init<<<(NODES + T - 1) / T, T, 0, stream>>>(dis, outdeg, NODES);
    mcp_deg_count<<<(NE + T - 1) / T, T, 0, stream>>>(src, dst, NE, dis, outdeg);
    mcp_deg_rsqrt<<<(NODES + T - 1) / T, T, 0, stream>>>(dis, NODES);
    hipMemsetAsync(As, 0, (size_t)NODES * KCLUS * sizeof(float), stream);

    // ---- pack weights (one-time, tiny) ----
    mcp_pack_b<<<packBlocks(FDIM, FDIM), T, 0, stream>>>(W1, W1p, FDIM, FDIM);
    mcp_pack_b<<<packBlocks(FDIM, FDIM), T, 0, stream>>>(W2, W2p, FDIM, FDIM);
    mcp_pack_b<<<packBlocks(FDIM, FDIM), T, 0, stream>>>(Wa1, Wa1p, FDIM, FDIM);
    mcp_pack_b<<<packBlocks(FDIM, KCLUS), T, 0, stream>>>(Wa2, Wa2p, FDIM, KCLUS);
    mcp_pack_b<<<packBlocks(FDIM, FDIM), T, 0, stream>>>(Wo, Wop, FDIM, FDIM);

    // ---- GCN layer 1: h1 = relu(aggregate(x@W1) + b1) ----
    mcp_cvt_f16<<<(NODES * FDIM / 4 + T - 1) / T, T, 0, stream>>>(x, F16A, NODES * FDIM);
    mcp_gemm_xw<<<gemmBlocks(NODES, FDIM), T, 0, stream>>>(F16A, W1p, nullptr, hw, nullptr,
                                                           NODES, FDIM, 0);
    mcp_selfloop<<<elemBlocks, T, 0, stream>>>(hw, dis, h1, NODES);
    mcp_edge_scatter<<<waveEdgeBlocks, T, 0, stream>>>(src, dst, dis, hw, h1, NE);
    mcp_bias_act<<<elemBlocks, T, 0, stream>>>(h1, b1, F16B, NODES, 1);

    // ---- GCN layer 2: h2 = aggregate(h1@W2) + b2 ----
    mcp_gemm_xw<<<gemmBlocks(NODES, FDIM), T, 0, stream>>>(F16B, W2p, nullptr, hw, nullptr,
                                                           NODES, FDIM, 0);
    mcp_selfloop<<<elemBlocks, T, 0, stream>>>(hw, dis, h2, NODES);
    mcp_edge_scatter<<<waveEdgeBlocks, T, 0, stream>>>(src, dst, dis, hw, h2, NE);
    mcp_bias_act<<<elemBlocks, T, 0, stream>>>(h2, b2, F16B, NODES, 0);   // F16B = h2h

    // ---- assignment: s = softmax(softmax(tanh(h2@Wa1+ba1)@Wa2+ba2)) ----
    mcp_gemm_xw<<<gemmBlocks(NODES, FDIM), T, 0, stream>>>(F16B, Wa1p, ba1, hw, F16A,
                                                           NODES, FDIM, 2);  // F16A = th
    mcp_gemm_xw<<<gemmBlocks(NODES, KCLUS), T, 0, stream>>>(F16A, Wa2p, ba2, s, nullptr,
                                                            NODES, KCLUS, 0);
    mcp_softmax2<<<waveNodeBlocks, T, 0, stream>>>(s, NODES);

    // ---- As = A @ s (edge-wise) + operand packing for pooling einsums ----
    mcp_as_scatter<<<waveEdgeBlocks, T, 0, stream>>>(src, dst, s, As, NE);
    mcp_pack_b<<<packBlocks(NODES, FDIM), T, 0, stream>>>(h2, F16A, NODES, FDIM); // F16A = h2p
    mcp_pack_b<<<packBlocks(NODES, KCLUS), T, 0, stream>>>(s, sp, NODES, KCLUS);
    mcp_pack_b<<<packBlocks(NODES, KCLUS), T, 0, stream>>>(As, Asp, NODES, KCLUS);
    mcp_transpose_s<<<(NODES * KCLUS + T - 1) / T, T, 0, stream>>>(s, sTh, NODES);

    // ---- pooling einsums (WMMA, K=4096 per batch) ----
    mcp_gemm_atb<<<(B_GRAPHS * 4 * (FDIM >> 4) + 7) / 8, T, 0, stream>>>(sTh, F16A, poolX,
                                                                         poolXh, FDIM);
    mcp_gemm_atb<<<(B_GRAPHS * 4 * (KCLUS >> 4) + 7) / 8, T, 0, stream>>>(sTh, Asp, adjRaw,
                                                                          nullptr, KCLUS);
    mcp_gemm_atb<<<(B_GRAPHS * 4 * (KCLUS >> 4) + 7) / 8, T, 0, stream>>>(sTh, sp, ssRaw,
                                                                          nullptr, KCLUS);

    // ---- losses + normalized pooled adjacency ----
    mcp_den<<<waveNodeBlocks, T, 0, stream>>>(s, outdeg, den, NODES);
    mcp_finalize<<<B_GRAPHS, T, 0, stream>>>(ssRaw, adjRaw, den, out + offAdj, out + offLoss);

    // ---- pooled_x = relu(poolX[512,128] @ Wo + bo) -> d_out ----
    mcp_gemm_xw<<<gemmBlocks(B_GRAPHS * KCLUS, FDIM), T, 0, stream>>>(
        poolXh, Wop, bo, out, nullptr, B_GRAPHS * KCLUS, FDIM, 1);
}
